// VQLayer_44882408243313
// MI455X (gfx1250) — compile-verified, matrix operand order checked by reading, not run on hardware
//
#include <hip/hip_runtime.h>
#include <stdint.h>

// Problem sizes (fixed by the reference).
#define B_TOT   65536
#define N_IN    1024
#define K_CODE  1024
#define E_DIM   256
#define TILE_M  32
#define LOG_K   6.9314718055994531f   // log(1024)

typedef __attribute__((ext_vector_type(16))) __bf16    v16bf;
typedef __attribute__((ext_vector_type(8)))  float     v8f;
typedef __attribute__((ext_vector_type(4)))  uint32_t  uv4;
typedef __attribute__((ext_vector_type(4)))  float     fv4;

union ABu { uv4 q[2]; v16bf v; };

// f32 -> bf16 (round to nearest even)
__device__ __forceinline__ uint32_t f2bf(float f) {
  uint32_t u = __float_as_uint(f);
  u += 0x7FFFu + ((u >> 16) & 1u);
  return u >> 16;
}

// monotone float <-> uint mapping (for packed max+argmax via u64 atomic max)
__device__ __forceinline__ uint32_t orderedF(float f) {
  uint32_t u = __float_as_uint(f);
  return (u & 0x80000000u) ? ~u : (u | 0x80000000u);
}
__device__ __forceinline__ float unorderedF(uint32_t h) {
  return (h & 0x80000000u) ? __uint_as_float(h ^ 0x80000000u)
                           : __uint_as_float(~h);
}

// ---- workspace layout (bytes) ----
#define WS_WBF_OFF   0u                                 // 1024*1024 bf16 = 2 MB
#define WS_AVGP_OFF  (2u * 1024u * 1024u)               // 1024 f32
#define WS_MIDX_OFF  (WS_AVGP_OFF + 4096u)              // 65536 i32

// ---- dynamic LDS layout (bytes) ----
#define SM_XS_OFF      0u                               // 32*1024 bf16 = 64 KB
#define SM_COLSUM_OFF  (TILE_M * N_IN * 2u)             // 1024 f32  = 4 KB
#define SM_ROWPK_OFF   (SM_COLSUM_OFF + K_CODE * 4u)    // 32 u64
#define SM_ROWSUM_OFF  (SM_ROWPK_OFF + TILE_M * 8u)     // 32 f32
#define SM_BYTES       (SM_ROWSUM_OFF + TILE_M * 4u)    // 70,016 B

__global__ __launch_bounds__(256) void zero_avgp(float* __restrict__ avgp) {
  avgp[blockIdx.x * 256 + threadIdx.x] = 0.0f;
}

__global__ __launch_bounds__(256) void wconv(const float* __restrict__ W,
                                             uint16_t* __restrict__ Wb) {
  int i = blockIdx.x * 256 + threadIdx.x;
  Wb[i] = (uint16_t)f2bf(W[i]);
}

// One WG = 32 rows of x, full K=1024 columns. 8 waves, each wave owns 128 cols
// and two 16-row M-tiles (each B-frag feeds two WMMAs). B-frags are manually
// software-pipelined so loads for tile t+1 are in flight during tile t's WMMAs.
__global__ __launch_bounds__(256) void vq_gemm_softmax(
    const float* __restrict__ x, const float* __restrict__ gumbel,
    const uint16_t* __restrict__ Wb, const float* __restrict__ b_enc,
    float* __restrict__ avgp, int* __restrict__ m_idx, float* __restrict__ out_m)
{
  extern __shared__ char smem[];
  uint16_t*           xs      = (uint16_t*)(smem + SM_XS_OFF);
  float*              colSum  = (float*)(smem + SM_COLSUM_OFF);
  unsigned long long* rowPack = (unsigned long long*)(smem + SM_ROWPK_OFF);
  float*              rowSum  = (float*)(smem + SM_ROWSUM_OFF);

  const int tid  = threadIdx.x;
  const int wave = tid >> 5;
  const int lane = tid & 31;
  const int cin  = lane & 15;   // N (and M for A-frag) position within a 16-tile
  const int grp  = lane >> 4;   // K half-group selector
  const int b0   = blockIdx.x * TILE_M;

  // stage + convert x tile to bf16 in LDS (vectorized: b128 load, b64 store)
  for (int i = tid * 4; i < TILE_M * N_IN; i += 256 * 4) {
    int r = i >> 10, c = i & (N_IN - 1);
    fv4 xv = *reinterpret_cast<const fv4*>(&x[(size_t)(b0 + r) * N_IN + c]);
    uint32_t p0 = f2bf(xv[0]) | (f2bf(xv[1]) << 16);
    uint32_t p1 = f2bf(xv[2]) | (f2bf(xv[3]) << 16);
    uint32_t* dst = reinterpret_cast<uint32_t*>(&xs[i]);
    dst[0] = p0; dst[1] = p1;
  }
  if (tid < TILE_M) { rowPack[tid] = 0ull; rowSum[tid] = 0.0f; }
  for (int i = tid; i < K_CODE; i += 256) colSum[i] = 0.0f;
  __syncthreads();

  v8f acc[2][8];
  #pragma unroll
  for (int mt = 0; mt < 2; ++mt)
    #pragma unroll
    for (int t = 0; t < 8; ++t)
      #pragma unroll
      for (int v = 0; v < 8; ++v) acc[mt][t][v] = 0.0f;

  // Per-lane base offsets.
  const uint16_t* wlane = Wb + (size_t)(wave * 128 + cin) * N_IN + grp * 8;
  const int aoff0 = cin * N_IN + grp * 8;          // M-tile 0 A-frag base
  const int aoff1 = (16 + cin) * N_IN + grp * 8;   // M-tile 1 A-frag base

  // ---- software-pipelined GEMM main loop over K=1024 in steps of 32 ----
  ABu a0c, a1c, bc;
  a0c.q[0] = *reinterpret_cast<const uv4*>(&xs[aoff0]);
  a0c.q[1] = *reinterpret_cast<const uv4*>(&xs[aoff0 + 16]);
  a1c.q[0] = *reinterpret_cast<const uv4*>(&xs[aoff1]);
  a1c.q[1] = *reinterpret_cast<const uv4*>(&xs[aoff1 + 16]);
  bc.q[0]  = *reinterpret_cast<const uv4*>(wlane);
  bc.q[1]  = *reinterpret_cast<const uv4*>(wlane + 16);

  for (int kk = 0; kk < N_IN; kk += 32) {
    ABu a0n, a1n;
    #pragma unroll
    for (int t = 0; t < 8; ++t) {
      // Prefetch B-frag for the next tile (t+1), or next K-step's tile 0.
      ABu bn;
      {
        const int tn = (t + 1) & 7;
        const int kn = (t == 7) ? (kk + 32) : kk;
        const uint16_t* wp = wlane + (size_t)tn * 16 * N_IN + kn;
        bn.q[0] = *reinterpret_cast<const uv4*>(wp);
        bn.q[1] = *reinterpret_cast<const uv4*>(wp + 16);
      }
      // Prefetch next K-step's A-frags while WMMAs drain.
      if (t == 6) {
        const int kn = kk + 32;
        a0n.q[0] = *reinterpret_cast<const uv4*>(&xs[aoff0 + kn]);
        a0n.q[1] = *reinterpret_cast<const uv4*>(&xs[aoff0 + kn + 16]);
        a1n.q[0] = *reinterpret_cast<const uv4*>(&xs[aoff1 + kn]);
        a1n.q[1] = *reinterpret_cast<const uv4*>(&xs[aoff1 + kn + 16]);
      }
      acc[0][t] = __builtin_amdgcn_wmma_f32_16x16x32_bf16(
          false, a0c.v, false, bc.v, (short)0, acc[0][t], false, false);
      acc[1][t] = __builtin_amdgcn_wmma_f32_16x16x32_bf16(
          false, a1c.v, false, bc.v, (short)0, acc[1][t], false, false);
      bc = bn;
    }
    a0c = a0n;
    a1c = a1n;
  }

  // ---- epilogue: scores = logits + bias + gumbel  (TAU == 1) ----
  // C layout: lane cin = column-in-tile; grp selects rows 0-7 vs 8-15; VGPR v = row offset
  #pragma unroll
  for (int mt = 0; mt < 2; ++mt)
    #pragma unroll
    for (int t = 0; t < 8; ++t) {
      const int c = wave * 128 + t * 16 + cin;
      const float be = b_enc[c];
      #pragma unroll
      for (int v = 0; v < 8; ++v) {
        const int r = mt * 16 + grp * 8 + v;
        acc[mt][t][v] = acc[mt][t][v] + be + gumbel[(size_t)(b0 + r) * K_CODE + c];
      }
    }

  // per-row max + argmax, packed into one u64 LDS atomic max per row
  #pragma unroll
  for (int mt = 0; mt < 2; ++mt)
    #pragma unroll
    for (int v = 0; v < 8; ++v) {
      float mx = acc[mt][0][v];
      int bc2 = wave * 128 + cin;
      #pragma unroll
      for (int t = 1; t < 8; ++t) {
        const int c = wave * 128 + t * 16 + cin;
        if (acc[mt][t][v] > mx) { mx = acc[mt][t][v]; bc2 = c; }
      }
      unsigned long long pk =
          ((unsigned long long)orderedF(mx) << 32) | (unsigned)bc2;
      atomicMax(&rowPack[mt * 16 + grp * 8 + v], pk);
    }
  __syncthreads();

  float rmax[2][8], psum[2][8];
  #pragma unroll
  for (int mt = 0; mt < 2; ++mt)
    #pragma unroll
    for (int v = 0; v < 8; ++v) {
      rmax[mt][v] = unorderedF((uint32_t)(rowPack[mt * 16 + grp * 8 + v] >> 32));
      psum[mt][v] = 0.0f;
    }
  #pragma unroll
  for (int mt = 0; mt < 2; ++mt)
    #pragma unroll
    for (int t = 0; t < 8; ++t)
      #pragma unroll
      for (int v = 0; v < 8; ++v) {
        float e = __expf(acc[mt][t][v] - rmax[mt][v]);
        acc[mt][t][v] = e;
        psum[mt][v] += e;
      }
  #pragma unroll
  for (int mt = 0; mt < 2; ++mt)
    #pragma unroll
    for (int v = 0; v < 8; ++v)
      atomicAdd(&rowSum[mt * 16 + grp * 8 + v], psum[mt][v]);
  __syncthreads();

  float rden[2][8];
  #pragma unroll
  for (int mt = 0; mt < 2; ++mt)
    #pragma unroll
    for (int v = 0; v < 8; ++v)
      rden[mt][v] = 1.0f / rowSum[mt * 16 + grp * 8 + v];

  // per-column sum of soft over this WG's 32 rows
  #pragma unroll
  for (int t = 0; t < 8; ++t) {
    const int c = wave * 128 + t * 16 + cin;
    float cs = 0.0f;
    #pragma unroll
    for (int mt = 0; mt < 2; ++mt)
      #pragma unroll
      for (int v = 0; v < 8; ++v) cs += acc[mt][t][v] * rden[mt][v];
    atomicAdd(&colSum[c], cs);
  }
  __syncthreads();

  // fold WG column sums into the global avg_p accumulator
  for (int i = tid; i < K_CODE; i += 256) atomicAdd(&avgp[i], colSum[i]);

  // argmax index output (m is the 2nd return; output buffer is f32)
  if (tid < TILE_M) {
    unsigned long long pk = rowPack[tid];
    int idx = (int)(uint32_t)(pk & 0xFFFFFFFFull);
    m_idx[b0 + tid] = idx;
    out_m[b0 + tid] = (float)idx;
  }
}

// z_q[b,:] = codebook[m[b],:]  (hard one-hot @ codebook == row gather)
__global__ __launch_bounds__(256) void zq_gather(
    const float* __restrict__ codebook, const int* __restrict__ m_idx,
    float* __restrict__ zq)
{
  const int b = blockIdx.x;
  const int e = threadIdx.x;
  const int m = m_idx[b];
  zq[(size_t)b * E_DIM + e] = codebook[(size_t)m * E_DIM + e];
}

__global__ __launch_bounds__(256) void diversity_kernel(
    const float* __restrict__ avgp, float* __restrict__ out)
{
  __shared__ float red[256];
  const int tid = threadIdx.x;
  float s = 0.0f;
  for (int i = tid; i < K_CODE; i += 256) {
    float p = avgp[i] * (1.0f / (float)B_TOT);
    s += p * (__logf(fmaxf(p, 1e-9f)) + LOG_K);
  }
  red[tid] = s;
  __syncthreads();
  for (int st = 128; st > 0; st >>= 1) {
    if (tid < st) red[tid] += red[tid + st];
    __syncthreads();
  }
  if (tid == 0) { out[0] = red[0]; out[1] = 0.0f; }
}

extern "C" void kernel_launch(void* const* d_in, const int* in_sizes, int n_in,
                              void* d_out, int out_size, void* d_ws, size_t ws_size,
                              hipStream_t stream) {
  (void)in_sizes; (void)n_in; (void)out_size; (void)ws_size;
  const float* x        = (const float*)d_in[0];
  const float* gumbel   = (const float*)d_in[1];
  const float* W_enc    = (const float*)d_in[2];
  const float* b_enc    = (const float*)d_in[3];
  const float* codebook = (const float*)d_in[4];

  uint16_t* Wb   = (uint16_t*)((char*)d_ws + WS_WBF_OFF);
  float*    avgp = (float*)   ((char*)d_ws + WS_AVGP_OFF);
  int*      midx = (int*)     ((char*)d_ws + WS_MIDX_OFF);

  float* out_zq = (float*)d_out;                       // [B, E]
  float* out_m  = out_zq + (size_t)B_TOT * E_DIM;      // [B]
  float* out_dv = out_m + B_TOT;                       // [diversity, zero]

  zero_avgp<<<K_CODE / 256, 256, 0, stream>>>(avgp);
  wconv<<<(K_CODE * N_IN) / 256, 256, 0, stream>>>(W_enc, Wb);
  vq_gemm_softmax<<<B_TOT / TILE_M, 256, SM_BYTES, stream>>>(
      x, gumbel, Wb, b_enc, avgp, midx, out_m);
  zq_gather<<<B_TOT, 256, 0, stream>>>(codebook, midx, out_zq);
  diversity_kernel<<<1, 256, 0, stream>>>(avgp, out_dv);
}
